// MSSSIM_8744553414957
// MI455X (gfx1250) — matrix operand order repaired and checked by reading, not compiled, not run on hardware
//
#include <hip/hip_runtime.h>

typedef __attribute__((ext_vector_type(16))) _Float16 v16h;
typedef __attribute__((ext_vector_type(8)))  _Float16 v8h;
typedef __attribute__((ext_vector_type(8)))  float    v8f;

#define NW 8              // waves per block (wave32)
#define C1F 1.0e-4f       // (0.01*L)^2
#define C2F 9.0e-4f       // (0.03*L)^2
#define GINV 0.26601157f  // 1 / sum_t exp(-(t-5)^2/4.5), t=0..10

// Normalized Gaussian tap for offset d; zero outside [0,10].
__device__ __forceinline__ float gtap(int d)
{
    float xx = (float)(d - 5);
    float v  = GINV * __expf(-xx * xx / 4.5f);
    return (d >= 0 && d < 11) ? v : 0.f;
}

// ---------------------------------------------------------------------------
// Fused SSIM level kernel: one wave per 16x16 output tile.
//
// Separable 11-tap Gaussian conv as Toeplitz WMMAs, one shared constant T:
//   horizontal: H(16x16)   = A_img(16x32) x T(32x16),  T[k][c] = g[k-c]
//   vertical  : outT(16x16)= A_Ht (16x32) x T(32x16)   (output transposed)
// The horizontal D layout maps 1:1 onto the vertical A-operand layout
// (a[e]=D0[e], a[e+8]=D1[e] per lane), so no cross-lane traffic is needed.
// Product planes (x^2, y^2, xy) are elementwise f16 products of the x/y
// A-operands, so only two LDS planes are staged, once, with one fence.
// ---------------------------------------------------------------------------
__global__ __launch_bounds__(NW * 32)
void ssim_level_kernel(const float* __restrict__ img1,
                       const float* __restrict__ img2,
                       int size, int nc, float* __restrict__ accum)
{
    __shared__ __align__(16) _Float16 plx[NW][1024];   // 32 rows x 32 halves, zero padded
    __shared__ __align__(16) _Float16 ply[NW][1024];

    const int lane = threadIdx.x & 31;
    const int w    = threadIdx.x >> 5;

    const int OW          = size - 10;
    const int tilesX      = (OW + 15) >> 4;
    const int tilesPerImg = tilesX * tilesX;
    const int total       = nc * tilesPerImg;

    int tile = blockIdx.x * NW + w;
    const bool liveTile = (tile < total);
    if (!liveTile) tile = 0;                            // keep EXEC full; mask later
    const int  img = tile / tilesPerImg;
    const int  rem = tile - img * tilesPerImg;
    const int  tyi = rem / tilesX;
    const int  txi = rem - tyi * tilesX;
    const int  y0  = tyi << 4, x0 = txi << 4;
    const long base = (long)img * size * size;

    // Shared Toeplitz B operand: lane holds col c=lane&15,
    // kb=(lane>>4)*8; element e <-> K=kb+e (e<8), K=kb+e+8 (e>=8); value g[K-c].
    v16h bh;
    {
        const int c = lane & 15, kb = (lane >> 4) * 8;
#pragma unroll
        for (int e = 0; e < 16; e++) bh[e] = (_Float16)gtap(kb + e + (e < 8 ? 0 : 8) - c);
    }

    // Zero both planes once (pad rows 26..31 / cols 26..31 stay zero).
    {
        unsigned* zx = (unsigned*)&plx[w][0];
        unsigned* zy = (unsigned*)&ply[w][0];
#pragma unroll
        for (int i = 0; i < 16; i++) { zx[i * 32 + lane] = 0u; zy[i * 32 + lane] = 0u; }
    }

    // Stage the 26x26 windows in f16: lane = column (edge-clamped),
    // coalesced global_load_b32 per row.
    const int gx = min(x0 + lane, size - 1);
#pragma unroll
    for (int r = 0; r < 26; r++) {
        int  gy  = min(y0 + r, size - 1);
        long off = base + (long)gy * size + gx;
        float x = img1[off];
        float y = img2[off];
        if (lane < 26) {
            plx[w][(r << 5) + lane] = (_Float16)x;
            ply[w][(r << 5) + lane] = (_Float16)y;
        }
    }
    // Same-wave DS ops are processed in order; just stop compiler motion.
    __builtin_amdgcn_wave_barrier();
    asm volatile("" ::: "memory");

    // A operands for x and y (rows 0..15 and 16..31; 26..31 are zero pad).
    const int arow = lane & 15;
    const int akb  = (lane >> 4) * 8;
    v16h ax0, ax1, ay0, ay1;
    {
        v8h lo = *(const v8h*)&plx[w][(arow << 5) + akb];
        v8h hi = *(const v8h*)&plx[w][(arow << 5) + akb + 16];
#pragma unroll
        for (int e = 0; e < 8; e++) { ax0[e] = lo[e]; ax0[e + 8] = hi[e]; }
    }
    {
        v8h lo = *(const v8h*)&plx[w][((arow + 16) << 5) + akb];
        v8h hi = *(const v8h*)&plx[w][((arow + 16) << 5) + akb + 16];
#pragma unroll
        for (int e = 0; e < 8; e++) { ax1[e] = lo[e]; ax1[e + 8] = hi[e]; }
    }
    {
        v8h lo = *(const v8h*)&ply[w][(arow << 5) + akb];
        v8h hi = *(const v8h*)&ply[w][(arow << 5) + akb + 16];
#pragma unroll
        for (int e = 0; e < 8; e++) { ay0[e] = lo[e]; ay0[e + 8] = hi[e]; }
    }
    {
        v8h lo = *(const v8h*)&ply[w][((arow + 16) << 5) + akb];
        v8h hi = *(const v8h*)&ply[w][((arow + 16) << 5) + akb + 16];
#pragma unroll
        for (int e = 0; e < 8; e++) { ay1[e] = lo[e]; ay1[e + 8] = hi[e]; }
    }

    // Product-plane A operands: elementwise packed f16 products.
    const v16h axx0 = ax0 * ax0, axx1 = ax1 * ax1;
    const v16h ayy0 = ay0 * ay0, ayy1 = ay1 * ay1;
    const v16h axy0 = ax0 * ay0, axy1 = ax1 * ay1;

    // One conv plane: 2 horizontal WMMAs + pack + 1 (transposed) vertical WMMA.
    auto conv = [&](const v16h& A0, const v16h& A1) -> v8f {
        v8f z{};
        v8f d0 = __builtin_amdgcn_wmma_f32_16x16x32_f16(false, A0, false, bh, (short)0, z, false, false);
        v8f d1 = __builtin_amdgcn_wmma_f32_16x16x32_f16(false, A1, false, bh, (short)0, z, false, false);
        v16h ah;
#pragma unroll
        for (int j = 0; j < 8; j++) { ah[j] = (_Float16)d0[j]; ah[j + 8] = (_Float16)d1[j]; }
        return __builtin_amdgcn_wmma_f32_16x16x32_f16(false, ah, false, bh, (short)0, z, false, false);
    };

    v8f D[5];
    D[0] = conv(ax0,  ax1);    // mu1^T
    D[1] = conv(ay0,  ay1);    // mu2^T
    D[2] = conv(axx0, axx1);   // E[x^2]^T
    D[3] = conv(ayy0, ayy1);   // E[y^2]^T
    D[4] = conv(axy0, axy1);   // E[xy]^T

    // Pointwise SSIM / CS (transposed layout: VGPR j -> column, lane -> row).
    float ssum = 0.f, csum = 0.f;
    const int mcol = (lane >> 4) * 8;
    const int nrow = lane & 15;
#pragma unroll
    for (int j = 0; j < 8; j++) {
        int  ox = x0 + mcol + j, oy = y0 + nrow;
        bool ok = liveTile && (oy < OW) && (ox < OW);
        float mu1 = D[0][j], mu2 = D[1][j];
        float ex2 = D[2][j], ey2 = D[3][j], exy = D[4][j];
        float mu1s = mu1 * mu1, mu2s = mu2 * mu2, mu12 = mu1 * mu2;
        float s1 = ex2 - mu1s, s2 = ey2 - mu2s, s12 = exy - mu12;
        float v1 = 2.f * s12 + C2F, v2 = s1 + s2 + C2F;
        float cs = v1 / v2;
        float ss = (2.f * mu12 + C1F) * v1 / ((mu1s + mu2s + C1F) * v2);
        if (ok) { ssum += ss; csum += cs; }
    }
#pragma unroll
    for (int m = 16; m >= 1; m >>= 1) {
        ssum += __shfl_xor(ssum, m, 32);
        csum += __shfl_xor(csum, m, 32);
    }
    if (lane == 0) {
        atomicAdd(&accum[0], ssum);
        atomicAdd(&accum[1], csum);
    }
}

// ---------------------------------------------------------------------------
// 2x2 average pool (stride 2) on both images of a level.
// ---------------------------------------------------------------------------
__global__ void avgpool2_kernel(const float* __restrict__ inA, const float* __restrict__ inB,
                                float* __restrict__ outA, float* __restrict__ outB,
                                int nc, int hin)
{
    const int  ho  = hin >> 1;
    const long n   = (long)nc * ho * ho;
    const long idx = (long)blockIdx.x * blockDim.x + threadIdx.x;
    if (idx >= n) return;
    int  img = (int)(idx / ((long)ho * ho));
    int  r   = (int)((idx / ho) % ho);
    int  c   = (int)(idx % ho);
    long b   = (long)img * hin * hin;
    long o0  = b + (long)(2 * r) * hin + 2 * c;
    long o1  = o0 + hin;
    outA[idx] = 0.25f * (inA[o0] + inA[o0 + 1] + inA[o1] + inA[o1 + 1]);
    outB[idx] = 0.25f * (inB[o0] + inB[o0 + 1] + inB[o1] + inB[o1 + 1]);
}

__global__ void zero_accum_kernel(float* a)
{
    if (threadIdx.x < 16) a[threadIdx.x] = 0.f;
}

__global__ void finalize_kernel(const float* __restrict__ acc, float* __restrict__ out)
{
    const float Wt[5] = {0.0448f, 0.2856f, 0.3001f, 0.2363f, 0.1333f};
    float res = 1.f;
#pragma unroll
    for (int l = 0; l < 5; l++) {
        int   s   = 512 >> l;
        int   ow  = s - 10;
        float cnt = (float)(48) * (float)ow * (float)ow;   // exact (< 2^24)
        float ms  = (acc[2 * l]     / cnt + 1.f) * 0.5f;
        float mc  = (acc[2 * l + 1] / cnt + 1.f) * 0.5f;
        res *= __powf((l < 4) ? mc : ms, Wt[l]);
    }
    out[0] = res;
}

// ---------------------------------------------------------------------------
extern "C" void kernel_launch(void* const* d_in, const int* in_sizes, int n_in,
                              void* d_out, int out_size, void* d_ws, size_t ws_size,
                              hipStream_t stream)
{
    const float* img1 = (const float*)d_in[0];
    const float* img2 = (const float*)d_in[1];
    float* acc = (float*)d_ws;
    float* p   = acc + 16;

    const int NC = 48;   // 16 batch * 3 channels, depthwise -> independent planes

    float* A[5]; float* B[5];
    A[0] = (float*)img1; B[0] = (float*)img2;
    size_t s1 = (size_t)NC * 256 * 256;
    size_t s2 = (size_t)NC * 128 * 128;
    size_t s3 = (size_t)NC * 64 * 64;
    size_t s4 = (size_t)NC * 32 * 32;
    A[1] = p;            B[1] = A[1] + s1;
    A[2] = B[1] + s1;    B[2] = A[2] + s2;
    A[3] = B[2] + s2;    B[3] = A[3] + s3;
    A[4] = B[3] + s3;    B[4] = A[4] + s4;

    hipLaunchKernelGGL(zero_accum_kernel, dim3(1), dim3(32), 0, stream, acc);

    for (int l = 0; l < 5; l++) {
        int sz = 512 >> l;
        if (l > 0) {
            int  hin = sz << 1;
            long n   = (long)NC * sz * sz;
            int  thr = 256;
            long blocks = (n + thr - 1) / thr;
            hipLaunchKernelGGL(avgpool2_kernel, dim3((unsigned)blocks), dim3(thr), 0, stream,
                               A[l - 1], B[l - 1], A[l], B[l], NC, hin);
        }
        int  ow     = sz - 10;
        int  tX     = (ow + 15) >> 4;
        long total  = (long)NC * tX * tX;
        long blocks = (total + NW - 1) / NW;
        hipLaunchKernelGGL(ssim_level_kernel, dim3((unsigned)blocks), dim3(NW * 32), 0, stream,
                           A[l], B[l], sz, NC, acc + 2 * l);
    }

    hipLaunchKernelGGL(finalize_kernel, dim3(1), dim3(1), 0, stream, acc, (float*)d_out);
}
